// CotLayer2d_22325240005125
// MI455X (gfx1250) — matrix operand order repaired and checked by reading, not compiled, not run on hardware
//
#include <hip/hip_runtime.h>
#include <hip/hip_bf16.h>
#include <math.h>
#include <stdint.h>

// ---------------- problem constants (match reference) ----------------
#define DIMC   128
#define HH     96
#define WW2    96
#define HWN    9216          // 96*96
#define NB     8
#define WC     144           // K*K*DIM/SP
#define ATTN   64

typedef __attribute__((ext_vector_type(16))) __bf16 v16bf;
typedef __attribute__((ext_vector_type(2)))  __bf16 v2bf;
typedef __attribute__((ext_vector_type(8)))  float  v8f;
typedef __attribute__((ext_vector_type(4)))  float  v4f;
typedef __attribute__((ext_vector_type(4)))  unsigned int u32x4;
typedef __attribute__((ext_vector_type(8)))  int  i32x8;
typedef __attribute__((ext_vector_type(4)))  int  i32x4;

// D = A(16x32 bf16) * B(32x16 bf16) + C(16x16 f32)
__device__ __forceinline__ v8f wmma_bf16(v16bf a, v16bf b, v8f c) {
  return __builtin_amdgcn_wmma_f32_16x16x32_bf16(
      /*neg_a=*/false, a, /*neg_b=*/false, b,
      /*c_mod=*/(short)0, c, /*reuse_a=*/false, /*reuse_b=*/false);
}

// gather one 16-bit A operand from LDS row [m][rowStride], K-tile base kt0.
// contiguous k order -> one 16-element (32B) run: k = kt0+hi*16+{0..15}, same as B.
__device__ __forceinline__ v16bf gather_a(const __bf16* sA, int mrow,
                                          int rowStride, int kt0, int hi) {
  v16bf av;
#pragma unroll
  for (int i = 0; i < 16; ++i) {
    int p  = i >> 1;
    int kk = (p << 1) + (i & 1) + (hi << 4);
    av[i] = sA[mrow * rowStride + kt0 + kk];
  }
  return av;
}

// pack 4 bf16 (2x v_cvt_pk-style pairs) and store as one 8-byte LDS write
__device__ __forceinline__ void store_pk4(__bf16* dst, __bf16 a, __bf16 b,
                                          __bf16 c, __bf16 d) {
  v2bf lo = {a, b}, hi = {c, d};
  uint2 u;
  u.x = __builtin_bit_cast(unsigned, lo);
  u.y = __builtin_bit_cast(unsigned, hi);
  *(uint2*)dst = u;
}

// -------- Tensor Data Mover: 1-row contiguous copy global(bf16) -> LDS -----
// D# group0: count=1 | lds_addr | global_addr[56:0] | type=2
// D# group1: data_size=2B, tensor_dim0=n, tensor_dim1=1, tile_dim0=n
// This toolchain exposes the 6-arg builtin: (g0, g1, g2, g3, g4, cpol).
__device__ __forceinline__ void tdm_load_1d(const __bf16* gsrc,
                                            unsigned lds_byte_addr, int nelem) {
  unsigned long long ga = (unsigned long long)(uintptr_t)gsrc;
  u32x4 g0;
  g0[0] = 1u;                                   // count=1, user mode
  g0[1] = lds_byte_addr;                        // lds_addr
  g0[2] = (unsigned)ga;                         // global_addr[31:0]
  g0[3] = (unsigned)((ga >> 32) & 0x01ffffffull) | (2u << 30); // [56:32]|type=2
  i32x8 g1;
  g1[0] = (int)(1u << 16);                      // data_size=1 -> 2 bytes
  g1[1] = (int)(((unsigned)nelem & 0xffffu) << 16);          // tensor_dim0 lo
  g1[2] = (int)((((unsigned)nelem >> 16) & 0xffffu) | (1u << 16)); // hi | td1=1
  g1[3] = (int)(((unsigned)nelem & 0xffffu) << 16);          // tile_dim0 = n
  g1[4] = 0;                                    // tile_dim1=0, tile_dim2=0
  g1[5] = (int)nelem;                           // tensor_dim0_stride
  g1[6] = 0; g1[7] = 0;
  i32x4 z; z[0] = 0; z[1] = 0; z[2] = 0; z[3] = 0;
  i32x8 z8; z8[0] = 0; z8[1] = 0; z8[2] = 0; z8[3] = 0;
  z8[4] = 0; z8[5] = 0; z8[6] = 0; z8[7] = 0;
  __builtin_amdgcn_tensor_load_to_lds(g0, g1, z, z, z8, 0);
}

// two-value wave32 + cross-wave block reduction (256 threads)
__device__ __forceinline__ void block_reduce2(float& s, float& ss) {
#pragma unroll
  for (int o = 16; o > 0; o >>= 1) {
    s  += __shfl_xor(s,  o, 32);
    ss += __shfl_xor(ss, o, 32);
  }
  __shared__ float red[16];
  int wid = threadIdx.x >> 5, lane = threadIdx.x & 31;
  if (lane == 0) { red[wid * 2] = s; red[wid * 2 + 1] = ss; }
  __syncthreads();
  if (threadIdx.x == 0) {
    float a = 0.f, b = 0.f;
    for (int i = 0; i < 8; ++i) { a += red[i * 2]; b += red[i * 2 + 1]; }
    s = a; ss = b;
  }
}

// ------------------- weight preparation (f32 -> bf16) ----------------------
__global__ __launch_bounds__(256) void cvt_bf16_kernel(
    const float* __restrict__ src, __bf16* __restrict__ dst, int n) {
  int i = blockIdx.x * 256 + threadIdx.x;
  if (i < n) dst[i] = (__bf16)src[i];
}

// key_w (oc, ic, t) -> bf16 (oc, t, ic): per-group block contiguous for TDM
__global__ __launch_bounds__(256) void keyw_permute_kernel(
    const float* __restrict__ src, __bf16* __restrict__ dst) {
  int i = blockIdx.x * 256 + threadIdx.x;   // 128*288
  int oc = i / 288, r = i - oc * 288;
  int t = r >> 5, ic = r & 31;
  dst[i] = (__bf16)src[(oc * 32 + ic) * 9 + t];
}

// ---------------------------------------------------------------------
// Per-batch GEMM: C[b][m][pix] = act( A[m][k] * B[b][k][pix] + bias )
// A: bf16 weights (row-major M x KD), loaded whole to LDS via TDM.
// B rows < split from B0, else B1 (concat(x,k)). 8 waves, 16 pixels each.
// ---------------------------------------------------------------------
template <int MT, int KD, bool RELU, bool BIAS>
__global__ __launch_bounds__(256) void gemm_wmma_kernel(
    const __bf16* __restrict__ Abf,
    const float* __restrict__ B0, size_t strideB0,
    const float* __restrict__ B1, size_t strideB1, int split,
    const float* __restrict__ bias, float* __restrict__ C) {
  const int b  = blockIdx.y;
  const int p0 = blockIdx.x * 128;
  const float* B0b = B0 + (size_t)b * strideB0;
  const float* B1b = B1 + (size_t)b * strideB1;
  float* Cb = C + (size_t)b * (MT * 16) * HWN;

  __shared__ alignas(32) __bf16 sA[MT * 16 * KD];
  __shared__ alignas(32) __bf16 sB[128 * 32];

  const int tid = threadIdx.x;
  const int wid = tid >> 5, lane = tid & 31;
  const int ln = lane & 15, hi = lane >> 4;
  const int pixb = (tid & 31) * 4;   // staging: 4-pixel quad
  const int kb   = (tid >> 5) * 4;   // staging: 4 k-rows

  // async TDM load of the full A matrix into LDS (one op, wave 0)
  if (tid < 32) {
    tdm_load_1d(Abf, (unsigned)(uintptr_t)&sA[0], MT * 16 * KD);
    __builtin_amdgcn_s_wait_tensorcnt(0);
  }

  v8f acc[MT];
#pragma unroll
  for (int m = 0; m < MT; ++m)
#pragma unroll
    for (int r = 0; r < 8; ++r) acc[m][r] = 0.f;

  __syncthreads();

  const int nk = KD >> 5;
  for (int kt = 0; kt < nk; ++kt) {
    // stage B tile (32 x 128) as bf16 [pix][k]; thread = 4 rows x 4 pixels:
    // 4x global_load_b128 + cvt-pairs + 4x ds_store_b64
    v4f rr[4];
#pragma unroll
    for (int r = 0; r < 4; ++r) {
      int row = kt * 32 + kb + r;
      const float* s = (row < split) ? (B0b + (size_t)row * HWN)
                                     : (B1b + (size_t)(row - split) * HWN);
      rr[r] = *(const v4f*)(s + p0 + pixb);
    }
#pragma unroll
    for (int j = 0; j < 4; ++j)
      store_pk4(&sB[(pixb + j) * 32 + kb], (__bf16)rr[0][j], (__bf16)rr[1][j],
                (__bf16)rr[2][j], (__bf16)rr[3][j]);
    if (kt + 1 < nk)  // gfx1250 global_prefetch_b8 for next K-slab
      __builtin_prefetch(B0b + (size_t)(kt * 32 + 32) * HWN + p0 + pixb, 0, 0);
    __syncthreads();

    v16bf bv = *(const v16bf*)(sB + (wid * 16 + ln) * 32 + hi * 16);
#pragma unroll
    for (int mt = 0; mt < MT; ++mt) {
      v16bf av = gather_a(sA, mt * 16 + ln, KD, kt * 32, hi);
      acc[mt] = wmma_bf16(av, bv, acc[mt]);
    }
    __syncthreads();
  }

  // epilogue: VGPR r -> M = mt*16 + hi*8 + r ; N = ln
#pragma unroll
  for (int mt = 0; mt < MT; ++mt)
#pragma unroll
    for (int r = 0; r < 8; ++r) {
      int m = mt * 16 + hi * 8 + r;
      float v = acc[mt][r];
      if (BIAS) v += bias[m];
      if (RELU) v = fmaxf(v, 0.f);
      Cb[(size_t)m * HWN + p0 + wid * 16 + ln] = v;
    }
}

// ---------------------------------------------------------------------
// Grouped 3x3 conv (groups=4, 32ch/group) + ReLU.
// Group weights (32x288 bf16, (m, t, ic) layout) loaded once via TDM.
// Per dy: each thread loads a 6-wide row segment for 4 input channels
// (1 aligned v4f + 2 masked edge scalars), assembles the three dx-shifted
// 32x128 bf16 tiles in LDS, then runs 3 WMMA k-steps.
// ---------------------------------------------------------------------
__global__ __launch_bounds__(256) void keyconv_kernel(
    const float* __restrict__ x, const __bf16* __restrict__ kwt,
    float* __restrict__ kout) {
  const int b = blockIdx.z, g = blockIdx.y;
  const int p0 = blockIdx.x * 128;
  const float* xb = x + (size_t)b * DIMC * HWN;

  __shared__ alignas(32) __bf16 sA[32 * 288];
  __shared__ alignas(32) __bf16 sB[3 * 128 * 32];   // one tile per dx

  const int tid = threadIdx.x;
  const int wid = tid >> 5, lane = tid & 31;
  const int ln = lane & 15, hi = lane >> 4;
  const int icq  = (tid >> 5) * 4;   // staging: 4 input-channel rows
  const int pixq = (tid & 31) * 4;   // staging: 4-pixel quad (same image row)
  const int h = (p0 + pixq) / WW2, w0 = (p0 + pixq) - h * WW2;

  if (tid < 32) {
    tdm_load_1d(kwt + (size_t)g * 32 * 288, (unsigned)(uintptr_t)&sA[0], 32 * 288);
    __builtin_amdgcn_s_wait_tensorcnt(0);
  }

  v8f acc[2];
#pragma unroll
  for (int m = 0; m < 2; ++m)
#pragma unroll
    for (int r = 0; r < 8; ++r) acc[m][r] = 0.f;

  __syncthreads();

  for (int dyi = 0; dyi < 3; ++dyi) {
    const int hh = h + dyi - 1;
    const float rm  = ((unsigned)hh < (unsigned)HH) ? 1.f : 0.f;
    const int hrow = (hh < 0 ? 0 : (hh > HH - 1 ? HH - 1 : hh)) * WW2;
    const float lm = (w0 > 0) ? rm : 0.f;            // left-edge mask
    const float em = (w0 < WW2 - 4) ? rm : 0.f;      // right-edge mask
    __bf16 L[4][6];
#pragma unroll
    for (int r = 0; r < 4; ++r) {
      const float* base = xb + (size_t)(g * 32 + icq + r) * HWN + hrow;
      float e0 = base[w0 > 0 ? w0 - 1 : 0] * lm;
      v4f m4 = *(const v4f*)(base + w0);
      m4 = m4 * rm;
      float e5 = base[w0 < WW2 - 4 ? w0 + 4 : WW2 - 1] * em;
      L[r][0] = (__bf16)e0;
      L[r][1] = (__bf16)m4[0]; L[r][2] = (__bf16)m4[1];
      L[r][3] = (__bf16)m4[2]; L[r][4] = (__bf16)m4[3];
      L[r][5] = (__bf16)e5;
    }
#pragma unroll
    for (int dxi = 0; dxi < 3; ++dxi)
#pragma unroll
      for (int j = 0; j < 4; ++j)
        store_pk4(&sB[dxi * 4096 + (pixq + j) * 32 + icq],
                  L[0][j + dxi], L[1][j + dxi], L[2][j + dxi], L[3][j + dxi]);
    __syncthreads();
#pragma unroll
    for (int dxi = 0; dxi < 3; ++dxi) {
      const int t = dyi * 3 + dxi;
      v16bf bv = *(const v16bf*)(sB + dxi * 4096 + (wid * 16 + ln) * 32 + hi * 16);
#pragma unroll
      for (int mt = 0; mt < 2; ++mt) {
        v16bf av = gather_a(sA, mt * 16 + ln, 288, t * 32, hi);
        acc[mt] = wmma_bf16(av, bv, acc[mt]);
      }
    }
    __syncthreads();
  }
#pragma unroll
  for (int mt = 0; mt < 2; ++mt)
#pragma unroll
    for (int r = 0; r < 8; ++r) {
      int oc = g * 32 + mt * 16 + hi * 8 + r;
      float v = fmaxf(acc[mt][r], 0.f);  // ReLU
      kout[((size_t)b * DIMC + oc) * HWN + p0 + wid * 16 + ln] = v;
    }
}

// GroupNorm stats: 16 groups of 9 contiguous channels, per batch
__global__ __launch_bounds__(256) void gn_stats_kernel(
    const float* __restrict__ wemb, float* __restrict__ gnst) {
  const int g = blockIdx.x, b = blockIdx.y;
  const float* base = wemb + ((size_t)b * WC + g * 9) * HWN;
  float s = 0.f, ss = 0.f;
  for (int q = threadIdx.x; q < (9 * HWN) / 4; q += 256) {
    v4f v = *(const v4f*)(base + q * 4);
#pragma unroll
    for (int j = 0; j < 4; ++j) { s += v[j]; ss += v[j] * v[j]; }
  }
  block_reduce2(s, ss);
  if (threadIdx.x == 0) {
    const float invN = 1.f / (9.f * HWN);
    float mean = s * invN;
    float var = ss * invN - mean * mean;
    gnst[(b * 16 + g) * 2] = mean;
    gnst[(b * 16 + g) * 2 + 1] = rsqrtf(var + 1e-5f);
  }
}

// y[b, cg*8+s, p] = sum_t GN(w[b, cg*9+t, p]) * v[b, cg*8+s, p+off(t)]
__global__ __launch_bounds__(256) void agg_kernel(
    const float* __restrict__ wemb, const float* __restrict__ vbuf,
    const float* __restrict__ gnst, const float* __restrict__ gn_g,
    const float* __restrict__ gn_b, float* __restrict__ y) {
  const int p4 = (blockIdx.x * 256 + threadIdx.x) * 4;
  const int cg = blockIdx.y, b = blockIdx.z;
  const int h = p4 / WW2, w0 = p4 - h * WW2;   // quad never crosses a row
  const float mean = gnst[(b * 16 + cg) * 2];
  const float rstd = gnst[(b * 16 + cg) * 2 + 1];
  v4f wn[9];
#pragma unroll
  for (int t = 0; t < 9; ++t) {
    int ch = cg * 9 + t;
    v4f raw = *(const v4f*)&wemb[((size_t)b * WC + ch) * HWN + p4];
    wn[t] = (raw - mean) * (rstd * gn_g[ch]) + gn_b[ch];
  }
#pragma unroll
  for (int s = 0; s < 8; ++s) {
    int c = cg * 8 + s;
    const float* vc = vbuf + ((size_t)b * DIMC + c) * HWN;
    v4f acc = {0.f, 0.f, 0.f, 0.f};
#pragma unroll
    for (int dyi = 0; dyi < 3; ++dyi) {
      int hh = h + dyi - 1;
      float rm = ((unsigned)hh < (unsigned)HH) ? 1.f : 0.f;
      int hrow = (hh < 0 ? 0 : (hh > HH - 1 ? HH - 1 : hh)) * WW2;
      float e0 = vc[hrow + (w0 > 0 ? w0 - 1 : 0)] * rm * (w0 > 0 ? 1.f : 0.f);
      v4f m4 = *(const v4f*)&vc[hrow + w0];
      m4 = m4 * rm;
      float e5 = vc[hrow + (w0 < WW2 - 4 ? w0 + 4 : WW2 - 1)] * rm *
                 (w0 < WW2 - 4 ? 1.f : 0.f);
      float Ls[6] = {e0, m4[0], m4[1], m4[2], m4[3], e5};
#pragma unroll
      for (int dxi = 0; dxi < 3; ++dxi) {
        v4f wv = wn[dyi * 3 + dxi];
#pragma unroll
        for (int j = 0; j < 4; ++j) acc[j] += wv[j] * Ls[j + dxi];
      }
    }
    *(v4f*)&y[((size_t)b * DIMC + c) * HWN + p4] = acc;
  }
}

// BatchNorm (training) stats per channel over (B, H, W)
__global__ __launch_bounds__(256) void bn_stats_kernel(
    const float* __restrict__ y, float* __restrict__ bnst) {
  const int c = blockIdx.x;
  float s = 0.f, ss = 0.f;
  for (int b = 0; b < NB; ++b) {
    const float* yp = y + ((size_t)b * DIMC + c) * HWN;
    for (int q = threadIdx.x; q < HWN / 4; q += 256) {
      v4f v = *(const v4f*)(yp + q * 4);
#pragma unroll
      for (int j = 0; j < 4; ++j) { s += v[j]; ss += v[j] * v[j]; }
    }
  }
  block_reduce2(s, ss);
  if (threadIdx.x == 0) {
    const float invN = 1.f / (float)(NB * HWN);
    float mean = s * invN;
    float var = ss * invN - mean * mean;
    bnst[c * 2] = mean;
    bnst[c * 2 + 1] = rsqrtf(var + 1e-5f);
  }
}

// y <- swish(BN(y)) in place; gap[b,c] = mean_p(y_final + k)
__global__ __launch_bounds__(256) void yfinal_gap_kernel(
    float* __restrict__ y, const float* __restrict__ k,
    const float* __restrict__ bnst, const float* __restrict__ bn_g,
    const float* __restrict__ bn_b, float* __restrict__ gap) {
  const int c = blockIdx.x, b = blockIdx.y;
  const float mean = bnst[c * 2], rstd = bnst[c * 2 + 1];
  const float gg = bn_g[c], bb = bn_b[c];
  float* yp = y + ((size_t)b * DIMC + c) * HWN;
  const float* kp = k + ((size_t)b * DIMC + c) * HWN;
  float s = 0.f, dummy = 0.f;
  for (int q = threadIdx.x; q < HWN / 4; q += 256) {
    v4f yv = *(const v4f*)(yp + q * 4);
    v4f kv = *(const v4f*)(kp + q * 4);
    v4f t = (yv - mean) * (rstd * gg) + bb;
    v4f yf;
#pragma unroll
    for (int j = 0; j < 4; ++j) {
      float tj = t[j];
      yf[j] = tj * (1.f / (1.f + expf(-tj)));  // swish
      s += yf[j] + kv[j];
    }
    *(v4f*)(yp + q * 4) = yf;
  }
  block_reduce2(s, dummy);
  if (threadIdx.x == 0) gap[b * DIMC + c] = s / (float)HWN;
}

// radix-2 SE: a1 = relu(W1*gap+b1); a2 = W2*a1+b2; softmax over radix pairs
__global__ __launch_bounds__(256) void se_kernel(
    const float* __restrict__ gap, const float* __restrict__ w1,
    const float* __restrict__ b1, const float* __restrict__ w2,
    const float* __restrict__ b2, float* __restrict__ attn) {
  const int b = blockIdx.x, tid = threadIdx.x;
  __shared__ float sg[DIMC], sa1[ATTN], sa2[2 * DIMC];
  if (tid < DIMC) sg[tid] = gap[b * DIMC + tid];
  __syncthreads();
  if (tid < ATTN) {
    float acc = b1[tid];
    for (int j = 0; j < DIMC; ++j) acc += w1[tid * DIMC + j] * sg[j];
    sa1[tid] = fmaxf(acc, 0.f);
  }
  __syncthreads();
  {
    float acc = b2[tid];
    for (int j = 0; j < ATTN; ++j) acc += w2[tid * ATTN + j] * sa1[j];
    sa2[tid] = acc;
  }
  __syncthreads();
  if (tid < DIMC) {
    float v0 = sa2[tid * 2], v1 = sa2[tid * 2 + 1];
    float m = fmaxf(v0, v1);
    float e0 = expf(v0 - m), e1 = expf(v1 - m);
    float inv = 1.f / (e0 + e1);
    attn[(b * DIMC + tid) * 2] = e0 * inv;
    attn[(b * DIMC + tid) * 2 + 1] = e1 * inv;
  }
}

// out = a0*y_final + a1*k  (4 pixels/thread)
__global__ __launch_bounds__(256) void mix_kernel(
    const float* __restrict__ y, const float* __restrict__ k,
    const float* __restrict__ attn, float* __restrict__ out) {
  int q = blockIdx.x * 256 + threadIdx.x;
  int bc = (q * 4) / HWN;  // b*128 + c (quad never crosses a channel)
  float a0 = attn[bc * 2], a1 = attn[bc * 2 + 1];
  v4f yv = *(const v4f*)(y + (size_t)q * 4);
  v4f kv = *(const v4f*)(k + (size_t)q * 4);
  v4f o = yv * a0 + kv * a1;
  *(v4f*)(out + (size_t)q * 4) = o;
}

// ---------------------------------------------------------------------
extern "C" void kernel_launch(void* const* d_in, const int* in_sizes, int n_in,
                              void* d_out, int out_size, void* d_ws, size_t ws_size,
                              hipStream_t stream) {
  const float* x     = (const float*)d_in[0];
  const float* key_w = (const float*)d_in[1];
  const float* e_w1  = (const float*)d_in[2];
  const float* e_w2  = (const float*)d_in[3];
  const float* e_b2  = (const float*)d_in[4];
  const float* gn_g  = (const float*)d_in[5];
  const float* gn_b  = (const float*)d_in[6];
  const float* c1_w  = (const float*)d_in[7];
  const float* bn_g  = (const float*)d_in[8];
  const float* bn_b  = (const float*)d_in[9];
  const float* se_w1 = (const float*)d_in[10];
  const float* se_b1 = (const float*)d_in[11];
  const float* se_w2 = (const float*)d_in[12];
  const float* se_b2 = (const float*)d_in[13];
  float* out = (float*)d_out;

  // workspace carve-up (~175 MB fp32 intermediates + bf16 weights)
  float* ws   = (float*)d_ws;
  float* kbuf = ws;                                   ws += (size_t)NB * DIMC * HWN;
  float* e1   = ws;                                   ws += (size_t)NB * 64   * HWN;
  float* wemb = ws;                                   ws += (size_t)NB * WC   * HWN;
  float* vbuf = ws;                                   ws += (size_t)NB * DIMC * HWN;
  float* ybuf = ws;                                   ws += (size_t)NB * DIMC * HWN;
  float* gnst = ws;                                   ws += NB * 16 * 2;
  float* bnst = ws;                                   ws += DIMC * 2;
  float* gapb = ws;                                   ws += NB * DIMC;
  float* attb = ws;                                   ws += NB * DIMC * 2;
  __bf16* wb     = (__bf16*)ws;
  __bf16* wb_e1  = wb;                                wb += 64 * 256;
  __bf16* wb_e2  = wb;                                wb += WC * 64;
  __bf16* wb_c1  = wb;                                wb += DIMC * DIMC;
  __bf16* wb_key = wb;                                wb += DIMC * 288;

  const size_t sX = (size_t)DIMC * HWN;   // per-batch stride of x / k / v / y
  const size_t sE = (size_t)64 * HWN;

  // 0. weight prep: f32 -> bf16 (key_w also permuted (oc,ic,t)->(oc,t,ic))
  cvt_bf16_kernel<<<(64 * 256) / 256, 256, 0, stream>>>(e_w1, wb_e1, 64 * 256);
  cvt_bf16_kernel<<<(WC * 64) / 256, 256, 0, stream>>>(e_w2, wb_e2, WC * 64);
  cvt_bf16_kernel<<<(DIMC * DIMC) / 256, 256, 0, stream>>>(c1_w, wb_c1, DIMC * DIMC);
  keyw_permute_kernel<<<(DIMC * 288) / 256, 256, 0, stream>>>(key_w, wb_key);

  // 1. k = relu(grouped 3x3 conv(x))                       [WMMA + TDM]
  keyconv_kernel<<<dim3(HWN / 128, 4, NB), 256, 0, stream>>>(x, wb_key, kbuf);
  // 2. e1 = relu(e_w1 * concat(x, k))  (64 x 256 GEMM)     [WMMA + TDM]
  gemm_wmma_kernel<4, 256, true, false><<<dim3(HWN / 128, NB), 256, 0, stream>>>(
      wb_e1, x, sX, kbuf, sX, DIMC, nullptr, e1);
  // 3. wemb = e_w2 * e1 + b2           (144 x 64 GEMM)     [WMMA + TDM]
  gemm_wmma_kernel<9, 64, false, true><<<dim3(HWN / 128, NB), 256, 0, stream>>>(
      wb_e2, e1, sE, e1, sE, 64, e_b2, wemb);
  // 4. v = c1_w * x                    (128 x 128 GEMM)    [WMMA + TDM]
  gemm_wmma_kernel<8, 128, false, false><<<dim3(HWN / 128, NB), 256, 0, stream>>>(
      wb_c1, x, sX, x, sX, DIMC, nullptr, vbuf);
  // 5-6. GroupNorm stats, then local per-pixel 3x3 aggregation
  gn_stats_kernel<<<dim3(16, NB), 256, 0, stream>>>(wemb, gnst);
  agg_kernel<<<dim3(HWN / 1024, 16, NB), 256, 0, stream>>>(wemb, vbuf, gnst,
                                                           gn_g, gn_b, ybuf);
  // 7-8. BatchNorm stats, swish in place + GAP(y + k)
  bn_stats_kernel<<<DIMC, 256, 0, stream>>>(ybuf, bnst);
  yfinal_gap_kernel<<<dim3(DIMC, NB), 256, 0, stream>>>(ybuf, kbuf, bnst,
                                                        bn_g, bn_b, gapb);
  // 9-10. SE radix softmax, final mix
  se_kernel<<<NB, 256, 0, stream>>>(gapb, se_w1, se_b1, se_w2, se_b2, attb);
  mix_kernel<<<(NB * DIMC * HWN) / 1024, 256, 0, stream>>>(ybuf, kbuf, attb, out);
}
